// OPFSurrogate_90683939488103
// MI455X (gfx1250) — matrix operand order repaired and checked by reading, not compiled, not run on hardware
//
#include <hip/hip_runtime.h>
#include <hip/hip_bf16.h>
#include <stdint.h>

// ---------------- problem constants (match reference) ----------------
#define NN 100000     // nodes
#define EE 3200000    // edges
#define FF 16         // in features
#define HH 64         // hidden
#define LL 3          // conv layers

typedef _Float16 v16h __attribute__((ext_vector_type(16)));
typedef _Float16 h8   __attribute__((ext_vector_type(8)));
typedef _Float16 h4   __attribute__((ext_vector_type(4)));
typedef float    v8f  __attribute__((ext_vector_type(8)));

// Fragment-packed B layout: Bp[((colt*2 + ks)*32 + lane)*16 + i], i=0..15
// where lane holds column n = colt*16 + (lane&15) and K = ks*32 + (lane>>4)*16 + i.
// Each lane's 16 halves are contiguous (32 B) -> two 16B loads, zero shuffling.

// ---------------------------------------------------------------------
// Pack / convert weights once:
//  - conv_W (f32 [3,64,64])  -> fragment-packed f16
//  - head W1 concat [64,128] -> fragment-packed f16 (3 quantile heads + conf)
//  - b1/W2/b2 concatenated (f32) for the head-finish kernel
// ---------------------------------------------------------------------
__global__ void k_pack(const float* __restrict__ convW, const float* __restrict__ qW1,
                       const float* __restrict__ cW1,  const float* __restrict__ qW2,
                       const float* __restrict__ cW2,  const float* __restrict__ qb1,
                       const float* __restrict__ cb1,  const float* __restrict__ qb2,
                       const float* __restrict__ cb2,
                       _Float16* __restrict__ convWp, _Float16* __restrict__ headW1p,
                       float* __restrict__ b1cat, float* __restrict__ W2cat,
                       float* __restrict__ b2cat) {
  int i = blockIdx.x * blockDim.x + threadIdx.x;
  if (i < LL * HH * HH) {                    // conv weights, fragment order
    int l = i >> 12, p = i & 4095;
    int i16 = p & 15, lane = (p >> 4) & 31, ks2 = (p >> 9) & 1, colt = (p >> 10) & 3;
    int k   = ks2 * 32 + (lane >> 4) * 16 + i16;
    int col = colt * 16 + (lane & 15);
    convWp[i] = (_Float16)convW[l * 4096 + k * HH + col];
    return;
  }
  int j = i - LL * HH * HH;
  if (j < HH * 128) {                        // fused head W1 [64,128], fragment order
    int i16 = j & 15, lane = (j >> 4) & 31, ks2 = (j >> 9) & 1, colt = (j >> 10) & 7;
    int k   = ks2 * 32 + (lane >> 4) * 16 + i16;
    int col = colt * 16 + (lane & 15);       // col = q*32+jj (q<3) | 96+jj (conf)
    float v;
    if (col < 96) { int q = col >> 5, jj = col & 31; v = qW1[(q * HH + k) * 32 + jj]; }
    else          { v = cW1[k * 32 + (col - 96)]; }
    headW1p[j] = (_Float16)v;
    return;
  }
  j -= HH * 128;
  if (j < 128) {
    float bv, wv;
    if (j < 96) { int q = j >> 5, jj = j & 31; bv = qb1[q * 32 + jj]; wv = qW2[q * 32 + jj]; }
    else        { bv = cb1[j - 96]; wv = cW2[j - 96]; }
    b1cat[j] = bv; W2cat[j] = wv; return;
  }
  j -= 128;
  if (j < 4) b2cat[j] = (j < 3) ? qb2[j] : cb2[0];
}

// ---------------------------------------------------------------------
// Encoder: one wave per node. lane handles features {lane, lane+32}.
// h = LayerNorm(ReLU(x @ W + b)) stored as f16 (WMMA A operand).
// ---------------------------------------------------------------------
__global__ void k_encoder(const float* __restrict__ x, const float* __restrict__ W,
                          const float* __restrict__ eb, const float* __restrict__ g,
                          const float* __restrict__ bb, _Float16* __restrict__ hh) {
  int node = blockIdx.x * (blockDim.x >> 5) + (threadIdx.x >> 5);
  if (node >= NN) return;
  int lane = threadIdx.x & 31;
  const float* xr = x + (size_t)node * FF;
  float v0 = eb[lane], v1 = eb[lane + 32];
#pragma unroll
  for (int k = 0; k < FF; ++k) {
    float xv = xr[k];
    v0 = fmaf(xv, W[k * HH + lane],      v0);
    v1 = fmaf(xv, W[k * HH + lane + 32], v1);
  }
  v0 = fmaxf(v0, 0.f); v1 = fmaxf(v1, 0.f);
  float s = v0 + v1;
#pragma unroll
  for (int off = 16; off > 0; off >>= 1) s += __shfl_xor(s, off, 32);
  float mu = s * (1.0f / 64.0f);
  float d0 = v0 - mu, d1 = v1 - mu;
  float q = d0 * d0 + d1 * d1;
#pragma unroll
  for (int off = 16; off > 0; off >>= 1) q += __shfl_xor(q, off, 32);
  float rs = rsqrtf(q * (1.0f / 64.0f) + 1e-5f);
  hh[(size_t)node * HH + lane]      = (_Float16)(d0 * rs * g[lane]      + bb[lane]);
  hh[(size_t)node * HH + lane + 32] = (_Float16)(d1 * rs * g[lane + 32] + bb[lane + 32]);
}

// ---------------------------------------------------------------------
// Degree with self-loop: deg = 1 + segment_sum(mask, dst)
// ---------------------------------------------------------------------
__global__ void k_init_deg(float* __restrict__ deg) {
  int i = blockIdx.x * blockDim.x + threadIdx.x;
  if (i < NN) deg[i] = 1.0f;
}

__global__ void k_degree(const long long* __restrict__ ei, const int* __restrict__ mask,
                         float* __restrict__ deg) {
  int e = blockIdx.x * blockDim.x + threadIdx.x;
  if (e >= EE) return;
  if (mask[e] > 0) unsafeAtomicAdd(&deg[(int)ei[(size_t)EE + e]], 1.0f);
}

__global__ void k_edge_norm(const long long* __restrict__ ei, const int* __restrict__ mask,
                            const float* __restrict__ deg, float* __restrict__ norm) {
  int e = blockIdx.x * blockDim.x + threadIdx.x;
  if (e >= EE) return;
  float w = 0.f;
  if (mask[e] > 0) {
    int s = (int)ei[e], d = (int)ei[(size_t)EE + e];
    w = rsqrtf(deg[s]) * rsqrtf(deg[d]);
  }
  norm[e] = w;
}

// ---------------------------------------------------------------------
// WMMA GEMM strip kernel: C[nRows, COLT*16](f32) = A[nRows,64](f16) @ B(f16, prepacked)
// One wave per 16-row strip: A fragment reused across COLT column tiles.
// K=64 -> 2 slabs; 2*COLT v_wmma_f32_16x16x32_f16 per wave.
// ---------------------------------------------------------------------
template <int COLT>
__global__ void k_gemm(const _Float16* __restrict__ A, const _Float16* __restrict__ Bp,
                       float* __restrict__ C, int nRows) {
  const int nCols = COLT * 16;
  int rowt = blockIdx.x * (blockDim.x >> 5) + (threadIdx.x >> 5);
  if (rowt >= (nRows >> 4)) return;           // wave-uniform: EXEC stays all-1s
  int lane = threadIdx.x & 31;
  int m  = lane & 15;                         // A row / B,C col within tile
  int hi = lane >> 4;

  v8f acc[COLT];
#pragma unroll
  for (int ct = 0; ct < COLT; ++ct) acc[ct] = (v8f){};

  const _Float16* arow = A + (size_t)(rowt * 16 + m) * HH + hi * 16;
#pragma unroll
  for (int ks = 0; ks < 2; ++ks) {
    // A 16x32 f16: lane = row m, halves = K {kb..kb+15}, kb = hi*16
    h8 a_lo = *(const h8*)(arow + ks * 32);
    h8 a_hi = *(const h8*)(arow + ks * 32 + 8);
    v16h a = __builtin_shufflevector(a_lo, a_hi,
                                     0, 1, 2, 3, 4, 5, 6, 7, 8, 9, 10, 11, 12, 13, 14, 15);
#pragma unroll
    for (int ct = 0; ct < COLT; ++ct) {
      // prepacked B fragment: 32 contiguous bytes per lane
      const h8* bp = (const h8*)(Bp + (size_t)(((ct * 2 + ks) * 32 + lane) * 16));
      h8 b_lo = bp[0];
      h8 b_hi = bp[1];
      v16h b = __builtin_shufflevector(b_lo, b_hi,
                                       0, 1, 2, 3, 4, 5, 6, 7, 8, 9, 10, 11, 12, 13, 14, 15);
      acc[ct] = __builtin_amdgcn_wmma_f32_16x16x32_f16(false, a, false, b, (short)0,
                                                       acc[ct], false, false);
    }
  }
  // C/D 16x16 f32: VGPR v -> row v + 8*hi, col = lane&15
#pragma unroll
  for (int ct = 0; ct < COLT; ++ct) {
    float* crow = C + (size_t)(rowt * 16 + hi * 8) * nCols + ct * 16 + m;
#pragma unroll
    for (int v = 0; v < 8; ++v) crow[(size_t)v * nCols] = acc[ct][v];
  }
}

// ---------------------------------------------------------------------
// agg = hw * self_norm (self loop, 1/deg); edge scatter adds on top.
// ---------------------------------------------------------------------
__global__ void k_agg_init(const float* __restrict__ hw, const float* __restrict__ deg,
                           float* __restrict__ agg) {
  unsigned t = (blockIdx.x * blockDim.x + threadIdx.x) * 4u;
  if (t >= (unsigned)NN * HH) return;
  unsigned n = t >> 6;
  float sn = 1.0f / deg[n];
  float4 v = *(const float4*)(hw + t);
  float4 r = make_float4(v.x * sn, v.y * sn, v.z * sn, v.w * sn);
  *(float4*)(agg + t) = r;
}

// Scatter: thread = (edge, 4-feature chunk). float4 gather + 4 HW f32 atomics.
__global__ void k_scatter(const long long* __restrict__ ei, const float* __restrict__ norm,
                          const float* __restrict__ hw, float* __restrict__ agg) {
  unsigned t = blockIdx.x * blockDim.x + threadIdx.x;
  unsigned e = t >> 4;
  if (e >= (unsigned)EE) return;
  float w = norm[e];
  if (w == 0.f) return;                        // outaged edges: no traffic
  int f4 = (t & 15) << 2;
  int s = (int)ei[e];
  int d = (int)ei[(size_t)EE + e];
  float4 v = *(const float4*)(hw + (size_t)s * HH + f4);
  float* dp = agg + (size_t)d * HH + f4;
  unsafeAtomicAdd(dp + 0, v.x * w);
  unsafeAtomicAdd(dp + 1, v.y * w);
  unsafeAtomicAdd(dp + 2, v.z * w);
  unsafeAtomicAdd(dp + 3, v.w * w);
}

// h_next = f16(ReLU(agg + bias)), 4 features / thread
__global__ void k_finish(const float* __restrict__ agg, const float* __restrict__ cb,
                         _Float16* __restrict__ hh) {
  unsigned t = (blockIdx.x * blockDim.x + threadIdx.x) * 4u;
  if (t >= (unsigned)NN * HH) return;
  int f = t & 63;
  float4 v = *(const float4*)(agg + t);
  h4 r;
  r.x = (_Float16)fmaxf(v.x + cb[f + 0], 0.f);
  r.y = (_Float16)fmaxf(v.y + cb[f + 1], 0.f);
  r.z = (_Float16)fmaxf(v.z + cb[f + 2], 0.f);
  r.w = (_Float16)fmaxf(v.w + cb[f + 3], 0.f);
  *(h4*)(hh + t) = r;
}

// ---------------------------------------------------------------------
// Finish heads: per node, 4x (bias + ReLU + 32-dot), sigmoid on head 3.
// hq is the [N,128] WMMA output of h @ headW1cat.
// ---------------------------------------------------------------------
__global__ void k_heads(const float* __restrict__ hq, const float* __restrict__ b1,
                        const float* __restrict__ W2, const float* __restrict__ b2,
                        float* __restrict__ out) {
  int n = blockIdx.x * blockDim.x + threadIdx.x;
  if (n >= NN) return;
  const float* row = hq + (size_t)n * 128;
#pragma unroll
  for (int q = 0; q < 4; ++q) {
    float s = b2[q];
#pragma unroll
    for (int j = 0; j < 32; ++j) {
      int c = q * 32 + j;
      s = fmaf(fmaxf(row[c] + b1[c], 0.f), W2[c], s);
    }
    out[(size_t)n * 4 + q] = (q < 3) ? s : (1.f / (1.f + expf(-s)));
  }
}

// ---------------------------------------------------------------------
extern "C" void kernel_launch(void* const* d_in, const int* in_sizes, int n_in,
                              void* d_out, int out_size, void* d_ws, size_t ws_size,
                              hipStream_t stream) {
  const float*      x      = (const float*)d_in[0];
  const long long*  ei     = (const long long*)d_in[1];   // int64 [2,E]
  const int*        omask  = (const int*)d_in[2];
  const float*      encW   = (const float*)d_in[3];
  const float*      encB   = (const float*)d_in[4];
  const float*      lnG    = (const float*)d_in[5];
  const float*      lnB    = (const float*)d_in[6];
  const float*      convW  = (const float*)d_in[7];
  const float*      convB  = (const float*)d_in[8];
  const float*      qW1    = (const float*)d_in[9];
  const float*      qb1    = (const float*)d_in[10];
  const float*      qW2    = (const float*)d_in[11];
  const float*      qb2    = (const float*)d_in[12];
  const float*      cW1    = (const float*)d_in[13];
  const float*      cb1    = (const float*)d_in[14];
  const float*      cW2    = (const float*)d_in[15];
  const float*      cb2    = (const float*)d_in[16];
  float* out = (float*)d_out;

  // ---- workspace layout (256B aligned) ----
  char* ws = (char*)d_ws;
  size_t o = 0;
  auto alloc = [&](size_t bytes) { void* p = ws + o; o += (bytes + 255) & ~(size_t)255; return p; };
  float*    hw      = (float*)   alloc((size_t)NN * 128 * sizeof(float));   // GEMM out (64/128 wide)
  _Float16* hh      = (_Float16*)alloc((size_t)NN * HH * sizeof(_Float16)); // f16 activations
  float*    agg     = (float*)   alloc((size_t)NN * HH * sizeof(float));
  float*    deg     = (float*)   alloc((size_t)NN * sizeof(float));
  float*    norm    = (float*)   alloc((size_t)EE * sizeof(float));
  _Float16* convWp  = (_Float16*)alloc((size_t)LL * HH * HH * sizeof(_Float16));
  _Float16* headW1p = (_Float16*)alloc((size_t)HH * 128 * sizeof(_Float16));
  float*    b1cat   = (float*)   alloc(128 * sizeof(float));
  float*    W2cat   = (float*)   alloc(128 * sizeof(float));
  float*    b2cat   = (float*)   alloc(4 * sizeof(float));
  (void)ws_size; (void)in_sizes; (void)n_in; (void)out_size;

  const int B = 256;
  // 1. weight pack / fragment reorder
  int packN = LL * HH * HH + HH * 128 + 128 + 4;
  k_pack<<<(packN + B - 1) / B, B, 0, stream>>>(convW, qW1, cW1, qW2, cW2, qb1, cb1,
                                                qb2, cb2, convWp, headW1p, b1cat, W2cat, b2cat);
  // 2. degrees + edge norms
  k_init_deg<<<(NN + B - 1) / B, B, 0, stream>>>(deg);
  k_degree<<<(EE + B - 1) / B, B, 0, stream>>>(ei, omask, deg);
  k_edge_norm<<<(EE + B - 1) / B, B, 0, stream>>>(ei, omask, deg, norm);
  // 3. encoder (one wave / node)
  k_encoder<<<NN / 8, B, 0, stream>>>(x, encW, encB, lnG, lnB, hh);
  // 4. GCN layers: one wave per 16-row strip
  int strips = NN / 16;                          // 6250
  int gemmBlocks = (strips + 7) / 8;
  for (int l = 0; l < LL; ++l) {
    k_gemm<4><<<gemmBlocks, B, 0, stream>>>(hh, convWp + (size_t)l * HH * HH, hw, NN);
    k_agg_init<<<((unsigned)NN * HH / 4 + B - 1) / B, B, 0, stream>>>(hw, deg, agg);
    k_scatter<<<((unsigned)EE * 16 + B - 1) / B, B, 0, stream>>>(ei, norm, hw, agg);
    k_finish<<<((unsigned)NN * HH / 4 + B - 1) / B, B, 0, stream>>>(agg, convB + (size_t)l * HH, hh);
  }
  // 5. fused heads: [N,64] @ [64,128] then per-node finish
  k_gemm<8><<<gemmBlocks, B, 0, stream>>>(hh, headW1p, hw, NN);
  k_heads<<<(NN + B - 1) / B, B, 0, stream>>>(hw, b1cat, W2cat, b2cat, out);
}
